// AIMv2Attention_33569464385834
// MI455X (gfx1250) — matrix operand order repaired and checked by reading, not compiled, not run on hardware
//
#include <hip/hip_runtime.h>
#include <hip/hip_bf16.h>

typedef __bf16 bf16;
typedef __attribute__((ext_vector_type(16))) __bf16 bf16x16;
typedef __attribute__((ext_vector_type(8)))  __bf16 bf16x8;
typedef __attribute__((ext_vector_type(8)))  float  f32x8;
typedef __attribute__((ext_vector_type(4)))  int    v4i;

#define WMMA_BF16(a, b, c) \
  __builtin_amdgcn_wmma_f32_16x16x32_bf16(false, (a), false, (b), (short)0, (c), false, false)

static constexpr int Bsz  = 4;
static constexpr int Nseq = 2048;
static constexpr int Cdim = 1024;
static constexpr int H    = 8;
static constexpr int Dh   = 128;
static constexpr int M1   = Bsz * Nseq;   // 8192 rows

// ---------------------------------------------------------------------------
// CDNA5 async global->LDS path (ASYNCcnt), with sync fallback if the builtins
// are not declared by this toolchain.  Builtin signature (from the round-2
// diagnostic): (v4i AS1*, v4i AS3*, imm offset, imm cpol).
// ---------------------------------------------------------------------------
#if __has_builtin(__builtin_amdgcn_global_load_async_to_lds_b128)
#define HAVE_ASYNC_LDS 1
#else
#define HAVE_ASYNC_LDS 0
#endif

__device__ __forceinline__ void copy16_g2l(const bf16* g, bf16* l) {
#if HAVE_ASYNC_LDS
  __builtin_amdgcn_global_load_async_to_lds_b128(
      (__attribute__((address_space(1))) v4i*)g,
      (__attribute__((address_space(3))) v4i*)l, 0, 0);
#else
  *(bf16x8*)l = *(const bf16x8*)g;
#endif
}

template <int N>
__device__ __forceinline__ void wait_async() {
#if HAVE_ASYNC_LDS
#if __has_builtin(__builtin_amdgcn_s_wait_asynccnt)
  __builtin_amdgcn_s_wait_asynccnt(N);
#else
  asm volatile("s_wait_asynccnt %0" ::"n"(N) : "memory");
#endif
#endif
}

// ---------------------------------------------------------------------------
// fp32 -> bf16 conversion (grid-stride)
// ---------------------------------------------------------------------------
__global__ void cvt_f32_bf16(const float* __restrict__ src, bf16* __restrict__ dst, int n) {
  int i = blockIdx.x * blockDim.x + threadIdx.x;
  int stride = gridDim.x * blockDim.x;
  for (; i < n; i += stride) dst[i] = (bf16)src[i];
}

// ---------------------------------------------------------------------------
// Fragment loaders (per-lane).  contiguous k order chosen here (A and B match):
//   A (16x32 bf16): lane L -> row M = L%16; elem i -> K = (L/16)*16 + i
//   B (32x16 bf16): lane L -> col N = L%16; elem i -> K = (L/16)*16 + i
// ---------------------------------------------------------------------------
__device__ __forceinline__ bf16x16 load_a_frag(const bf16* __restrict__ row, int k0, int half) {
  bf16x8 lo = *(const bf16x8*)(row + k0 + half * 16);
  bf16x8 hi = *(const bf16x8*)(row + k0 + half * 16 + 8);
  bf16x16 r;
#pragma unroll
  for (int i = 0; i < 8; ++i) { r[i] = lo[i]; r[i + 8] = hi[i]; }
  return r;
}
__device__ __forceinline__ bf16x16 load_a_frag_lds(const bf16* row, int k0, int half) {
  bf16x8 lo = *(const bf16x8*)(row + k0 + half * 16);
  bf16x8 hi = *(const bf16x8*)(row + k0 + half * 16 + 8);
  bf16x16 r;
#pragma unroll
  for (int i = 0; i < 8; ++i) { r[i] = lo[i]; r[i + 8] = hi[i]; }
  return r;
}
__device__ __forceinline__ bf16x16 load_b_frag(const bf16* __restrict__ p) {
  return *(const bf16x16*)p;
}
__device__ __forceinline__ bf16x16 load_b_frag_lds(const bf16* p) {
  return *(const bf16x16*)p;
}

// ---------------------------------------------------------------------------
// QKV GEMM: qkv[m,o] = sum_k X[m,k] * Wqkv[o,k]   (M=8192, O=3072, K=1024)
// Block = 128 threads = 4 waves (2x2), wave tile 64x64, block tile 128x128.
// Epilogue scatters into Q [B,H,N,D] (pre-scaled by 1/sqrt(D)), K [B,H,N,D],
// and V transposed [B,H,D,N] so all attention fragments load contiguously.
// ---------------------------------------------------------------------------
__global__ __launch_bounds__(128) void gemm_qkv(const bf16* __restrict__ X,
                                                const bf16* __restrict__ W,
                                                bf16* __restrict__ Qb,
                                                bf16* __restrict__ Kb,
                                                bf16* __restrict__ Vt) {
  const int lane = threadIdx.x & 31;
  const int wave = threadIdx.x >> 5;
  const int half = lane >> 4, l16 = lane & 15;
  const int m0 = blockIdx.x * 128 + (wave >> 1) * 64;
  const int o0 = blockIdx.y * 128 + (wave & 1) * 64;

  f32x8 acc[4][4] = {};
  for (int k0 = 0; k0 < Cdim; k0 += 32) {
    if (k0 + 32 < Cdim) {
      __builtin_prefetch(X + (size_t)(m0 + l16) * Cdim + k0 + 32, 0, 1);
      __builtin_prefetch(W + (size_t)(o0 + l16) * Cdim + k0 + 32, 0, 1);
    }
    bf16x16 a[4], b[4];
#pragma unroll
    for (int t = 0; t < 4; ++t)
      a[t] = load_a_frag(X + (size_t)(m0 + t * 16 + l16) * Cdim, k0, half);
#pragma unroll
    for (int u = 0; u < 4; ++u)
      b[u] = load_b_frag(W + (size_t)(o0 + u * 16 + l16) * Cdim + k0 + half * 16);
#pragma unroll
    for (int t = 0; t < 4; ++t)
#pragma unroll
      for (int u = 0; u < 4; ++u)
        acc[t][u] = WMMA_BF16(a[t], b[u], acc[t][u]);
  }

  const float qscale = 0.08838834764831845f;  // 1/sqrt(128)
#pragma unroll
  for (int t = 0; t < 4; ++t)
#pragma unroll
    for (int u = 0; u < 4; ++u)
#pragma unroll
      for (int r = 0; r < 8; ++r) {
        int m = m0 + t * 16 + half * 8 + r;
        int o = o0 + u * 16 + l16;
        float v = acc[t][u][r];
        int bb = m >> 11;            // m / 2048
        int n  = m & (Nseq - 1);
        int which = o >> 10;         // 0=q 1=k 2=v
        int c = o & (Cdim - 1);
        int h = c >> 7;
        int d = c & (Dh - 1);
        int bh = bb * H + h;
        if (which == 0)
          Qb[((size_t)bh * Nseq + n) * Dh + d] = (bf16)(v * qscale);
        else if (which == 1)
          Kb[((size_t)bh * Nseq + n) * Dh + d] = (bf16)v;
        else
          Vt[((size_t)bh * Dh + d) * Nseq + n] = (bf16)v;
      }
}

// ---------------------------------------------------------------------------
// Flash attention: grid (Nseq/64, H, B), block 128 (4 waves, 16 q-rows each).
// K/V chunks staged in LDS once per block via CDNA5 async global->LDS copies,
// software-pipelined: K double-buffered one chunk ahead; V issued at the top
// of the iteration and waited only before P*V.  Rows padded +8 bf16 so ds
// fragment reads are bank-conflict free.  P tiles bounce through LDS to go
// from C-layout to A-layout.
// ---------------------------------------------------------------------------
static constexpr int KP = Dh + 8;   // 136: padded K row (d-major)
static constexpr int VP = 64 + 8;   // 72:  padded V row (key-major)
static constexpr int PP = 64 + 8;   // 72:  padded P row

__global__ __launch_bounds__(128) void attn_fwd(const bf16* __restrict__ Qb,
                                                const bf16* __restrict__ Kb,
                                                const bf16* __restrict__ Vt,
                                                bf16* __restrict__ Yo) {
  __shared__ bf16 sK[2][64 * KP];   // [key][d]   2 x 17408 B
  __shared__ bf16 sV[Dh * VP];      // [d][key]   18432 B
  __shared__ bf16 sP[4][16 * PP];   // per-wave P  9216 B
  const int tid  = threadIdx.x;
  const int lane = tid & 31;
  const int wave = tid >> 5;
  const int half = lane >> 4, l16 = lane & 15;
  const int b = blockIdx.z, h = blockIdx.y;
  const int bh = b * H + h;
  const int q0 = blockIdx.x * 64 + wave * 16;

  const bf16* Kbase = Kb + (size_t)bh * Nseq * Dh;     // rows contiguous
  const bf16* Vbase = Vt + (size_t)bh * Dh * Nseq;     // [d][n]

  // Chunk copy helpers: 64x128 K chunk = 1024 16B pieces; 128x64 V chunk = 1024.
  auto issue_k = [&](int kc, int buf) {
#pragma unroll
    for (int c = 0; c < 8; ++c) {
      int q  = tid + c * 128;       // piece index
      int ky = q >> 4, dq = q & 15; // row, 8-elem column piece
      copy16_g2l(Kbase + (size_t)(kc + ky) * Dh + dq * 8, &sK[buf][ky * KP + dq * 8]);
    }
  };
  auto issue_v = [&](int kc) {
#pragma unroll
    for (int c = 0; c < 8; ++c) {
      int q = tid + c * 128;
      int d = q >> 3, kq = q & 7;
      copy16_g2l(Vbase + (size_t)d * Nseq + kc + kq * 8, &sV[d * VP + kq * 8]);
    }
  };

  // Preload Q fragments (16 rows x 128 D -> 4 A-frags across K-steps of 32)
  bf16x16 aq[4];
  const bf16* qrow = Qb + ((size_t)bh * Nseq + q0 + l16) * Dh;
#pragma unroll
  for (int ds = 0; ds < 4; ++ds) aq[ds] = load_a_frag(qrow, ds * 32, half);

  f32x8 oacc[8] = {};
  float m_s[8], l_s[8];
#pragma unroll
  for (int r = 0; r < 8; ++r) { m_s[r] = -1e30f; l_s[r] = 0.f; }

  issue_k(0, 0);  // prologue: K chunk 0 in flight

  const int nchunks = Nseq / 64;
  for (int i = 0; i < nchunks; ++i) {
    const int cur = i & 1;
    const bool has_next = (i + 1) < nchunks;
    issue_v(i * 64);
    if (has_next) issue_k((i + 1) * 64, cur ^ 1);

    // K_i must be resident (V_i + K_{i+1} may still be in flight)
    if (has_next) wait_async<16>(); else wait_async<8>();
    __syncthreads();

    // S = Q K^T (scale folded into Q); B-frags from padded LDS K tile
    f32x8 s[4] = {};
#pragma unroll
    for (int j = 0; j < 4; ++j) {
      const bf16* krow = &sK[cur][(j * 16 + l16) * KP];
#pragma unroll
      for (int ds = 0; ds < 4; ++ds) {
        bf16x16 bk = load_b_frag_lds(krow + ds * 32 + half * 16);
        s[j] = WMMA_BF16(aq[ds], bk, s[j]);
      }
    }

    // Online softmax: row M = r + half*8 lives in lanes {half*16 .. half*16+15}
#pragma unroll
    for (int r = 0; r < 8; ++r) {
      float v = fmaxf(fmaxf(s[0][r], s[1][r]), fmaxf(s[2][r], s[3][r]));
#pragma unroll
      for (int off = 1; off < 16; off <<= 1) v = fmaxf(v, __shfl_xor(v, off, 32));
      float nm = fmaxf(m_s[r], v);
      float alpha = __expf(m_s[r] - nm);
      m_s[r] = nm;
      float acc = 0.f;
#pragma unroll
      for (int j = 0; j < 4; ++j) {
        float p = __expf(s[j][r] - nm);
        s[j][r] = p;
        acc += p;
      }
#pragma unroll
      for (int off = 1; off < 16; off <<= 1) acc += __shfl_xor(acc, off, 32);
      l_s[r] = l_s[r] * alpha + acc;
#pragma unroll
      for (int dt = 0; dt < 8; ++dt) oacc[dt][r] *= alpha;
    }

    // Stage P (bf16) in per-wave LDS tile (C-layout -> memory row-major)
#pragma unroll
    for (int j = 0; j < 4; ++j)
#pragma unroll
      for (int r = 0; r < 8; ++r)
        sP[wave][(r + half * 8) * PP + j * 16 + l16] = (bf16)s[j][r];

    // V_i must be resident before P*V (K_{i+1} may still be in flight)
    if (has_next) wait_async<8>(); else wait_async<0>();
    __syncthreads();

    // O += P V   (V tile key-contiguous per d row)
#pragma unroll
    for (int ks = 0; ks < 2; ++ks) {
      bf16x16 ap = load_a_frag_lds(&sP[wave][l16 * PP], ks * 32, half);
#pragma unroll
      for (int dt = 0; dt < 8; ++dt) {
        bf16x16 bv = load_b_frag_lds(&sV[(dt * 16 + l16) * VP + ks * 32 + half * 16]);
        oacc[dt] = WMMA_BF16(ap, bv, oacc[dt]);
      }
    }
    __syncthreads();  // everyone done reading sK[cur], sV, sP
  }

  // Normalize and write back [B, N, C] (heads re-interleaved)
#pragma unroll
  for (int dt = 0; dt < 8; ++dt)
#pragma unroll
    for (int r = 0; r < 8; ++r) {
      int n = q0 + r + half * 8;
      float v = oacc[dt][r] / l_s[r];
      Yo[((size_t)b * Nseq + n) * Cdim + h * Dh + dt * 16 + l16] = (bf16)v;
    }
}

// ---------------------------------------------------------------------------
// Output projection: out[m,o] = sum_k A[m,k] * Wproj[o,k]  (8192x1024x1024),
// fp32 result straight to d_out.
// ---------------------------------------------------------------------------
__global__ __launch_bounds__(128) void gemm_proj(const bf16* __restrict__ X,
                                                 const bf16* __restrict__ W,
                                                 float* __restrict__ Y) {
  const int lane = threadIdx.x & 31;
  const int wave = threadIdx.x >> 5;
  const int half = lane >> 4, l16 = lane & 15;
  const int m0 = blockIdx.x * 128 + (wave >> 1) * 64;
  const int o0 = blockIdx.y * 128 + (wave & 1) * 64;

  f32x8 acc[4][4] = {};
  for (int k0 = 0; k0 < Cdim; k0 += 32) {
    if (k0 + 32 < Cdim) {
      __builtin_prefetch(X + (size_t)(m0 + l16) * Cdim + k0 + 32, 0, 1);
      __builtin_prefetch(W + (size_t)(o0 + l16) * Cdim + k0 + 32, 0, 1);
    }
    bf16x16 a[4], b[4];
#pragma unroll
    for (int t = 0; t < 4; ++t)
      a[t] = load_a_frag(X + (size_t)(m0 + t * 16 + l16) * Cdim, k0, half);
#pragma unroll
    for (int u = 0; u < 4; ++u)
      b[u] = load_b_frag(W + (size_t)(o0 + u * 16 + l16) * Cdim + k0 + half * 16);
#pragma unroll
    for (int t = 0; t < 4; ++t)
#pragma unroll
      for (int u = 0; u < 4; ++u)
        acc[t][u] = WMMA_BF16(a[t], b[u], acc[t][u]);
  }

#pragma unroll
  for (int t = 0; t < 4; ++t)
#pragma unroll
    for (int u = 0; u < 4; ++u)
#pragma unroll
      for (int r = 0; r < 8; ++r) {
        int m = m0 + t * 16 + half * 8 + r;
        int o = o0 + u * 16 + l16;
        Y[(size_t)m * Cdim + o] = acc[t][u][r];
      }
}

// ---------------------------------------------------------------------------
// Host-side orchestration
// ---------------------------------------------------------------------------
extern "C" void kernel_launch(void* const* d_in, const int* in_sizes, int n_in,
                              void* d_out, int out_size, void* d_ws, size_t ws_size,
                              hipStream_t stream) {
  const float* x     = (const float*)d_in[0];  // [B, N, C]
  const float* wqkv  = (const float*)d_in[1];  // [3C, C]
  const float* wproj = (const float*)d_in[2];  // [C, C]
  float* out = (float*)d_out;                  // [B, N, C]

  char* ws = (char*)d_ws;
  size_t off = 0;
  auto carve = [&](size_t bytes) {
    void* p = ws + off;
    off += (bytes + 255) & ~(size_t)255;
    return p;
  };
  bf16* xb  = (bf16*)carve((size_t)M1 * Cdim * sizeof(bf16));       // x in bf16
  bf16* wqb = (bf16*)carve((size_t)3 * Cdim * Cdim * sizeof(bf16)); // Wqkv bf16
  bf16* wpb = (bf16*)carve((size_t)Cdim * Cdim * sizeof(bf16));     // Wproj bf16
  bf16* Qb  = (bf16*)carve((size_t)M1 * Cdim * sizeof(bf16));       // [B,H,N,D]
  bf16* Kb  = (bf16*)carve((size_t)M1 * Cdim * sizeof(bf16));       // [B,H,N,D]
  bf16* Vt  = (bf16*)carve((size_t)M1 * Cdim * sizeof(bf16));       // [B,H,D,N]
  bf16* Ao  = (bf16*)carve((size_t)M1 * Cdim * sizeof(bf16));       // attn out [B,N,C]

  cvt_f32_bf16<<<4096, 256, 0, stream>>>(x, xb, M1 * Cdim);
  cvt_f32_bf16<<<2048, 256, 0, stream>>>(wqkv, wqb, 3 * Cdim * Cdim);
  cvt_f32_bf16<<<1024, 256, 0, stream>>>(wproj, wpb, Cdim * Cdim);

  gemm_qkv<<<dim3(M1 / 128, (3 * Cdim) / 128), 128, 0, stream>>>(xb, wqb, Qb, Kb, Vt);
  attn_fwd<<<dim3(Nseq / 64, H, Bsz), 128, 0, stream>>>(Qb, Kb, Vt, Ao);
  gemm_proj<<<dim3(M1 / 128, Cdim / 128), 128, 0, stream>>>(Ao, wpb, out);
}
